// DecoderAttention_86320252715495
// MI455X (gfx1250) — matrix operand order repaired and checked by reading, not compiled, not run on hardware
//
#include <hip/hip_runtime.h>
#include <hip/hip_bf16.h>
#include <math.h>

// ---------------------------------------------------------------------------
// Problem dimensions (fixed by the reference) — compile-time so strides fold
// into instruction immediates instead of v_add_nc_u64 chains.
// ---------------------------------------------------------------------------
#define BB   2
#define TT   2048
#define DD   1024
#define HH   16
#define HD   64
#define MM   (BB * TT)     // 4096
#define N3   (3 * DD)      // 3072

// ---------------------------------------------------------------------------
// Types for CDNA5 WMMA (wave32): D(16x16 f32) = A(16x32 bf16) x B(32x16 bf16) + C
// ---------------------------------------------------------------------------
typedef __bf16 bf16;
typedef __attribute__((ext_vector_type(16))) __bf16 bf16x16;
typedef __attribute__((ext_vector_type(8)))  __bf16 bf16x8;
typedef __attribute__((ext_vector_type(8)))  float  f32x8;
typedef __attribute__((address_space(3))) bf16 lds_bf16;

static __device__ __forceinline__ f32x8 wmma_bf16(bf16x16 a, bf16x16 b, f32x8 c) {
    return __builtin_amdgcn_wmma_f32_16x16x32_bf16(
        /*neg_a=*/false, a, /*neg_b=*/false, b,
        /*c_mod=*/(short)0, c, /*reuse_a=*/false, /*reuse_b=*/false);
}

// A-matrix 16x32 bf16 fragment (row-major source, leading dim ld elements).
// Lane l: row r=l&15, half hf=l>>4; elems 0..7 = A[r][hf*8+..], 8..15 = A[r][16+hf*8+..]
static __device__ __forceinline__ bf16x16 load_a_frag(const bf16* base, int ld, int lane) {
    const int r = lane & 15, hf = lane >> 4;
    const bf16* p = base + (size_t)r * ld + hf * 8;
    union { bf16x16 v; bf16x8 h[2]; } u;
    u.h[0] = *(const bf16x8*)(p);
    u.h[1] = *(const bf16x8*)(p + 16);
    return u.v;
}

// B-matrix 32x16 bf16 fragment from an N x K row-major source
// (computes sum_k A[m,k]*Bsrc[n,k]). Lane l: col n=l&15, hf=l>>4; elems 0..7 = Bsrc[n][hf*8+..], 8..15 = Bsrc[n][16+hf*8+..]
static __device__ __forceinline__ bf16x16 load_b_frag(const bf16* base, int ld, int lane) {
    const int n = lane & 15, hf = lane >> 4;
    const bf16* p = base + (size_t)n * ld + hf * 8;
    union { bf16x16 v; bf16x8 h[2]; } u;
    u.h[0] = *(const bf16x8*)(p);
    u.h[1] = *(const bf16x8*)(p + 16);
    return u.v;
}

static __device__ __forceinline__ float wave32_sum(float v) {
#pragma unroll
    for (int m = 16; m >= 1; m >>= 1) v += __shfl_xor(v, m, 32);
    return v;
}
static __device__ __forceinline__ float half16_max(float v) {
#pragma unroll
    for (int m = 8; m >= 1; m >>= 1) v = fmaxf(v, __shfl_xor(v, m, 32));
    return v;
}
static __device__ __forceinline__ float half16_sum(float v) {
#pragma unroll
    for (int m = 8; m >= 1; m >>= 1) v += __shfl_xor(v, m, 32);
    return v;
}

// ---------------------------------------------------------------------------
// Kernel 1: fp32 -> bf16 elementwise convert
// ---------------------------------------------------------------------------
__global__ void k_f2bf(const float* __restrict__ in, bf16* __restrict__ out, size_t n) {
    size_t i = (size_t)blockIdx.x * blockDim.x + threadIdx.x;
    if (i < n) out[i] = (bf16)in[i];
}

// ---------------------------------------------------------------------------
// Kernel 2: row RMS-norm (over D=1024) -> bf16. One block per row.
// ---------------------------------------------------------------------------
__global__ __launch_bounds__(256)
void k_rmsnorm_rows(const float* __restrict__ x, bf16* __restrict__ y) {
    const int row = blockIdx.x;
    const float* xp = x + (size_t)row * DD;
    bf16* yp = y + (size_t)row * DD;
    float ss = 0.f;
    for (int c = threadIdx.x; c < DD; c += 256) { float v = xp[c]; ss += v * v; }
    ss = wave32_sum(ss);
    __shared__ float red[8];
    if ((threadIdx.x & 31) == 0) red[threadIdx.x >> 5] = ss;
    __syncthreads();
    float tot = 0.f;
#pragma unroll
    for (int w = 0; w < 8; ++w) tot += red[w];
    const float r = rsqrtf(tot / (float)DD + 1e-6f);
    for (int c = threadIdx.x; c < DD; c += 256) yp[c] = (bf16)(xp[c] * r);
}

// ---------------------------------------------------------------------------
// Kernel 3: WMMA GEMM  C[M,N] = A[M,K] * Bsrc[N,K]^T   (bf16 in, fp32 out)
// K compile-time. Block = 256 threads = 8 waves.
// Block tile 64(M) x 512(N); wave tile 64 x 64 (4x4 WMMA tiles).
// Shared A tile (64x32, 4KB) staged into LDS with async global->LDS copies
// (ASYNCcnt), double-buffered.
// ---------------------------------------------------------------------------
template <int KD>
__global__ __launch_bounds__(256)
void k_gemm_bf16(const bf16* __restrict__ A, const bf16* __restrict__ Bsrc,
                 float* __restrict__ C, int N) {
    constexpr int KSTEPS = KD / 32;
    const int lane = threadIdx.x & 31;
    const int wave = threadIdx.x >> 5;
    const int m0 = blockIdx.x * 64;
    const int n0 = blockIdx.y * 512 + wave * 64;

    __shared__ __align__(16) bf16 sA[2][64 * 32];

    // Async staging: thread tid copies 16 bytes. row sr = tid>>2, col sc = (tid&3)*8.
    const int tid = threadIdx.x;
    const int sr = tid >> 2;
    const int sc = (tid & 3) * 8;
    const unsigned ldsOff0 = (unsigned)(size_t)(lds_bf16*)(&sA[0][sr * 32 + sc]);
    const int gOffBase = ((m0 + sr) * KD + sc) * 2;   // byte offset into A

    auto stage = [&](int bufsel, int k0) {
        unsigned l = ldsOff0 + (unsigned)bufsel * (64 * 32 * 2);
        int voff = gOffBase + k0 * 2;
        asm volatile("global_load_async_to_lds_b128 %0, %1, %2"
                     :: "v"(l), "v"(voff), "s"(A) : "memory");
    };

    f32x8 acc[4][4] = {};

    stage(0, 0);
    for (int j = 0; j < KSTEPS; ++j) {
        const int k0 = j * 32;
        if (j + 1 < KSTEPS) {
            stage((j + 1) & 1, k0 + 32);
            asm volatile("s_wait_asynccnt 0x1" ::: "memory");  // current buffer done
        } else {
            asm volatile("s_wait_asynccnt 0x0" ::: "memory");
        }
        __syncthreads();

        const bf16* cur = &sA[j & 1][0];
        bf16x16 af[4];
#pragma unroll
        for (int mt = 0; mt < 4; ++mt)
            af[mt] = load_a_frag(cur + mt * 16 * 32, 32, lane);
#pragma unroll
        for (int nt = 0; nt < 4; ++nt) {
            bf16x16 b = load_b_frag(Bsrc + (size_t)(n0 + nt * 16) * KD + k0, KD, lane);
#pragma unroll
            for (int mt = 0; mt < 4; ++mt)
                acc[mt][nt] = wmma_bf16(af[mt], b, acc[mt][nt]);
        }
        __syncthreads();   // all waves done reading before buffer reuse
    }

    const int col = lane & 15, hf = lane >> 4;
#pragma unroll
    for (int mt = 0; mt < 4; ++mt)
#pragma unroll
        for (int nt = 0; nt < 4; ++nt)
#pragma unroll
            for (int i = 0; i < 8; ++i) {
                const int row = m0 + mt * 16 + hf * 8 + i;
                C[(size_t)row * N + n0 + nt * 16 + col] = acc[mt][nt][i];
            }
}

// ---------------------------------------------------------------------------
// Kernel 4: per-head q/k RMS-norm + scale + RoPE; v transpose.
// One wave per (b,t,head). Lane d handles elements d and d+32.
// qkv fp32: [B*T, 3*H*64]; q head i at col i*64, k at (H+i)*64, v at (2H+i)*64.
// Outputs: Q,K as [b,h,T,64] bf16 ; V as [b,h,64,T] bf16 (transposed).
// ---------------------------------------------------------------------------
__global__ __launch_bounds__(32)
void k_qk_prep(const float* __restrict__ qkv, const float* __restrict__ cosT,
               const float* __restrict__ sinT, const float* __restrict__ temp,
               bf16* __restrict__ Q, bf16* __restrict__ Kc, bf16* __restrict__ Vt) {
    const int idx = blockIdx.x;
    const int head = idx % HH;
    const int t = (idx / HH) % TT;
    const int b = idx / (HH * TT);
    const int lane = threadIdx.x;
    const size_t rowOff = (size_t)(b * TT + t) * (3 * HH * HD);
    const float* qp = qkv + rowOff + (size_t)head * HD;
    const float* kp = qkv + rowOff + (size_t)(HH + head) * HD;
    const float* vp = qkv + rowOff + (size_t)(2 * HH + head) * HD;

    const float s1 = sqrtf(temp[lane] + 1.0f);
    const float s2 = sqrtf(temp[lane + 32] + 1.0f);
    const float c = cosT[(size_t)t * 32 + lane];
    const float s = sinT[(size_t)t * 32 + lane];

    const size_t qkBase = ((size_t)(b * HH + head) * TT + t) * HD;

    {   // q
        float x1 = qp[lane], x2 = qp[lane + 32];
        float ss = wave32_sum(x1 * x1 + x2 * x2);
        float r = rsqrtf(ss / 64.0f + 1e-6f);
        float y1 = x1 * r * s1, y2 = x2 * r * s2;
        Q[qkBase + lane]      = (bf16)(y1 * c - y2 * s);
        Q[qkBase + lane + 32] = (bf16)(y1 * s + y2 * c);
    }
    {   // k
        float x1 = kp[lane], x2 = kp[lane + 32];
        float ss = wave32_sum(x1 * x1 + x2 * x2);
        float r = rsqrtf(ss / 64.0f + 1e-6f);
        float y1 = x1 * r * s1, y2 = x2 * r * s2;
        Kc[qkBase + lane]      = (bf16)(y1 * c - y2 * s);
        Kc[qkBase + lane + 32] = (bf16)(y1 * s + y2 * c);
    }
    {   // v -> [b,h,hd,T]
        const size_t vBase = (size_t)(b * HH + head) * HD * TT;
        Vt[vBase + (size_t)lane * TT + t]        = (bf16)vp[lane];
        Vt[vBase + (size_t)(lane + 32) * TT + t] = (bf16)vp[lane + 32];
    }
}

// ---------------------------------------------------------------------------
// Kernel 5: causal flash attention. One wave per (16 q-rows, b*h).
// S = Q K^T via WMMA (hd=64 -> 2 K-steps); online softmax in base-2 domain
// (exp2 folds the hd^-1/2 scale and log2e into one constant mul); P staged
// through LDS (C-layout -> A-layout); P@V via WMMA with V^T as the B operand.
// K-loop split: bulk blocks strictly below the diagonal skip all causal
// masking; only the <=2 diagonal blocks pay the cmp/cndmask cost.
// ---------------------------------------------------------------------------
__global__ __launch_bounds__(32)
void k_attn(const bf16* __restrict__ Q, const bf16* __restrict__ Kc,
            const bf16* __restrict__ Vt, bf16* __restrict__ Out) {
    const int lane = threadIdx.x;
    const int qt = blockIdx.x;
    const int bh = blockIdx.y;
    const int b = bh / HH, head = bh % HH;
    const int m0 = qt * 16;
    const int col = lane & 15, hf = lane >> 4;

    const bf16* qbase = Q  + (size_t)bh * TT * HD;
    const bf16* kbase = Kc + (size_t)bh * TT * HD;
    const bf16* vbase = Vt + (size_t)bh * HD * TT;

    __shared__ __align__(16) bf16 P[16 * 32];

    const bf16x16 qf0 = load_a_frag(qbase + (size_t)m0 * HD,      HD, lane);
    const bf16x16 qf1 = load_a_frag(qbase + (size_t)m0 * HD + 32, HD, lane);

    f32x8 acc[4] = {};
    float mrow[8], lrow[8];
#pragma unroll
    for (int i = 0; i < 8; ++i) { mrow[i] = -INFINITY; lrow[i] = 0.f; }

    // hd^-1/2 * log2(e): scores scaled straight into the exp2 domain.
    const float SC = 0.125f * 1.44269504088896340736f;

    auto process = [&](int c0, bool MASKED) {
        f32x8 s[2];
#pragma unroll
        for (int t = 0; t < 2; ++t) {
            bf16x16 b0 = load_b_frag(kbase + (size_t)(c0 + t * 16) * HD,      HD, lane);
            bf16x16 b1 = load_b_frag(kbase + (size_t)(c0 + t * 16) * HD + 32, HD, lane);
            f32x8 z = {};
            z = wmma_bf16(qf0, b0, z);
            z = wmma_bf16(qf1, b1, z);
            s[t] = z;
        }
        float mnew[8];
#pragma unroll
        for (int i = 0; i < 8; ++i) {
            float v0 = s[0][i] * SC;
            float v1 = s[1][i] * SC;
            if (MASKED) {
                const int rg = m0 + hf * 8 + i;
                if (c0 + col > rg)      v0 = -INFINITY;
                if (c0 + 16 + col > rg) v1 = -INFINITY;
            }
            s[0][i] = v0; s[1][i] = v1;
            float mx = half16_max(fmaxf(v0, v1));
            mnew[i] = fmaxf(mrow[i], mx);
        }
        float alpha[8];
#pragma unroll
        for (int i = 0; i < 8; ++i) {
            alpha[i] = exp2f(mrow[i] - mnew[i]);
            mrow[i] = mnew[i];
        }
#pragma unroll
        for (int i = 0; i < 8; ++i) {
            float p0 = exp2f(s[0][i] - mnew[i]);
            float p1 = exp2f(s[1][i] - mnew[i]);
            P[(hf * 8 + i) * 32 + col]      = (bf16)p0;
            P[(hf * 8 + i) * 32 + 16 + col] = (bf16)p1;
            lrow[i] = lrow[i] * alpha[i] + half16_sum(p0 + p1);
        }
#pragma unroll
        for (int t = 0; t < 4; ++t)
#pragma unroll
            for (int i = 0; i < 8; ++i) acc[t][i] *= alpha[i];

        __builtin_amdgcn_wave_barrier();
        asm volatile("s_wait_dscnt 0x0" ::: "memory");   // P visible across lanes

        const bf16x16 pf = load_a_frag(P, 32, lane);
#pragma unroll
        for (int t = 0; t < 4; ++t) {
            bf16x16 vf = load_b_frag(vbase + (size_t)(t * 16) * TT + c0, TT, lane);
            acc[t] = wmma_bf16(pf, vf, acc[t]);
        }
        __builtin_amdgcn_wave_barrier();
        asm volatile("s_wait_dscnt 0x0" ::: "memory");   // done with P before overwrite
    };

    // Bulk: blocks with c0+31 <= m0 are fully unmasked.
    const int nFull = m0 / 32;
    for (int j = 0; j < nFull; ++j) process(j * 32, false);
    // Diagonal tail: masked blocks.
    for (int c0 = nFull * 32; c0 <= m0 + 15; c0 += 32) process(c0, true);

    float inv[8];
#pragma unroll
    for (int i = 0; i < 8; ++i) inv[i] = __builtin_amdgcn_rcpf(lrow[i]);
#pragma unroll
    for (int t = 0; t < 4; ++t)
#pragma unroll
        for (int i = 0; i < 8; ++i) {
            const int row = m0 + hf * 8 + i;
            const float o = acc[t][i] * inv[i];
            Out[((size_t)(b * TT + row)) * DD + head * HD + t * 16 + col] = (bf16)o;
        }
}

// ---------------------------------------------------------------------------
// Kernel 6: final residual + post-scale. out = rsqrt(mean(o^2)+1) * (input + o)
// ---------------------------------------------------------------------------
__global__ __launch_bounds__(256)
void k_final(const float* __restrict__ o, const float* __restrict__ inp,
             float* __restrict__ out) {
    const int row = blockIdx.x;
    const float* op = o + (size_t)row * DD;
    const float* ip = inp + (size_t)row * DD;
    float* yp = out + (size_t)row * DD;
    float ss = 0.f;
    for (int c = threadIdx.x; c < DD; c += 256) { float v = op[c]; ss += v * v; }
    ss = wave32_sum(ss);
    __shared__ float red[8];
    if ((threadIdx.x & 31) == 0) red[threadIdx.x >> 5] = ss;
    __syncthreads();
    float tot = 0.f;
#pragma unroll
    for (int w = 0; w < 8; ++w) tot += red[w];
    const float ps = rsqrtf(tot / (float)DD + 1.0f);
    for (int c = threadIdx.x; c < DD; c += 256) yp[c] = ps * (ip[c] + op[c]);
}

// ---------------------------------------------------------------------------
// Host launcher
// ---------------------------------------------------------------------------
extern "C" void kernel_launch(void* const* d_in, const int* in_sizes, int n_in,
                              void* d_out, int out_size, void* d_ws, size_t ws_size,
                              hipStream_t stream) {
    const float* input = (const float*)d_in[0];   // [B,T,D]
    const float* cosT  = (const float*)d_in[1];   // [T,32]
    const float* sinT  = (const float*)d_in[2];   // [T,32]
    const float* wqkv  = (const float*)d_in[3];   // [3D,D]
    const float* temp  = (const float*)d_in[4];   // [64]
    const float* wo    = (const float*)d_in[5];   // [D,D]
    float* out = (float*)d_out;

    char* ws = (char*)d_ws;
    size_t off = 0;
    auto alloc = [&](size_t bytes) -> void* {
        void* p = ws + off;
        off = (off + bytes + 255) & ~(size_t)255;
        return p;
    };
    bf16*  xnorm   = (bf16*)alloc((size_t)MM * DD * sizeof(bf16));
    bf16*  wqkv_bf = (bf16*)alloc((size_t)N3 * DD * sizeof(bf16));
    bf16*  wo_bf   = (bf16*)alloc((size_t)DD * DD * sizeof(bf16));
    float* qkv     = (float*)alloc((size_t)MM * N3 * sizeof(float));
    bf16*  Qb      = (bf16*)alloc((size_t)MM * DD * sizeof(bf16));
    bf16*  Kb      = (bf16*)alloc((size_t)MM * DD * sizeof(bf16));
    bf16*  Vb      = (bf16*)alloc((size_t)MM * DD * sizeof(bf16));
    bf16*  attnO   = (bf16*)alloc((size_t)MM * DD * sizeof(bf16));
    float* oproj   = (float*)alloc((size_t)MM * DD * sizeof(float));
    (void)ws_size; (void)n_in; (void)in_sizes; (void)out_size;

    {   // weights fp32 -> bf16
        size_t n = (size_t)N3 * DD;
        k_f2bf<<<(unsigned)((n + 255) / 256), 256, 0, stream>>>(wqkv, wqkv_bf, n);
        size_t n2 = (size_t)DD * DD;
        k_f2bf<<<(unsigned)((n2 + 255) / 256), 256, 0, stream>>>(wo, wo_bf, n2);
    }
    // input RMS norm -> bf16
    k_rmsnorm_rows<<<MM, 256, 0, stream>>>(input, xnorm);
    // QKV GEMM: [M,3072] = xnorm[M,1024] * Wqkv[3072,1024]^T
    k_gemm_bf16<DD><<<dim3(MM / 64, N3 / 512), 256, 0, stream>>>(xnorm, wqkv_bf, qkv, N3);
    // per-head q/k rmsnorm + scale + rope; v transpose
    k_qk_prep<<<BB * TT * HH, 32, 0, stream>>>(qkv, cosT, sinT, temp, Qb, Kb, Vb);
    // causal flash attention
    k_attn<<<dim3(TT / 16, BB * HH), 32, 0, stream>>>(Qb, Kb, Vb, attnO);
    // O projection: [M,1024] = attnO[M,1024] * Wo[1024,1024]^T
    k_gemm_bf16<DD><<<dim3(MM / 64, DD / 512), 256, 0, stream>>>(attnO, wo_bf, oproj, DD);
    // residual + post-scale
    k_final<<<MM, 256, 0, stream>>>(oproj, input, out);
}
